// WindowAttention_18124761989753
// MI455X (gfx1250) — matrix operand order repaired and checked by reading, not compiled, not run on hardware
//
#include <hip/hip_runtime.h>
#include <hip/hip_bf16.h>

// ---------------------------------------------------------------------------
// Window attention for MI455X (gfx1250), wave32 + WMMA f16->f32.
//   B=256 windows, N=256 tokens, C=384, 12 heads, head_dim=32 (== WMMA K!)
// 3 kernels: qkv(+bias+l2norm+scale) -> attention(softmax fused) -> proj.
// Intermediates staged in d_ws as f16:  q | k | v | attn_out  (4 x 48 MiB).
// This revision: software-pipelined fragment loads (overlap ds/global loads
// with WMMA instead of s_wait_dscnt 0 before every MMA), paired b32 LDS fills,
// and a K-permuted P/V layout so softmax writeback is 2x ds_store_b128
// instead of 16x ds_store_b16 per row.
// ---------------------------------------------------------------------------

#define B_WIN   256
#define DIM     384
#define NTOK    256
#define NHEAD   12
#define HDIM    32
#define SCALE_F 0.17677669529663689f   // 32^-0.5

typedef __attribute__((ext_vector_type(16))) _Float16 v16h;
typedef __attribute__((ext_vector_type(8)))  _Float16 v8h;
typedef __attribute__((ext_vector_type(2)))  _Float16 v2h;
typedef __attribute__((ext_vector_type(8)))  float    v8f;

__device__ __forceinline__ v8f vzero8() {
    v8f z;
#pragma unroll
    for (int i = 0; i < 8; ++i) z[i] = 0.0f;
    return z;
}

// D = A*B + C  (16x16x32, f16 in, f32 acc)
__device__ __forceinline__ v8f wmma_f16(v16h a, v16h b, v8f c) {
    return __builtin_amdgcn_wmma_f32_16x16x32_f16(false, a, false, b,
                                                  (short)0, c, false, false);
}

// A-matrix fragment (16x32 f16). K order used by this code (not the ISA table order):
//  lane half hl = lane>>4, m = lane&15.
//  element i = K { 16*hl + i } (contiguous, matching the B fragments).
// rowk = &A[m][kbase]  (row-major, contiguous K).  16B-aligned chunks.
__device__ __forceinline__ v16h load_frag_a(const _Float16* rowk, int hl) {
    const v8h lo = *(const v8h*)(rowk + 16 * hl);
    const v8h hi = *(const v8h*)(rowk + 16 * hl + 8);
    v16h f;
#pragma unroll
    for (int i = 0; i < 8; ++i) { f[i] = lo[i]; f[i + 8] = hi[i]; }
    return f;
}

// B-matrix fragment (32x16 f16). lanes 0-15: K=0..15, lanes 16-31: K=16..31,
// element i = K (16*(lane>>4) + i).  p = &B^T[n][kbase + 16*(lane>>4)]
// where B^T row n holds the K-contiguous data for output column n.
__device__ __forceinline__ v16h load_frag_b(const _Float16* p) {
    const v8h lo = *(const v8h*)(p);
    const v8h hi = *(const v8h*)(p + 8);
    v16h f;
#pragma unroll
    for (int i = 0; i < 8; ++i) { f[i] = lo[i]; f[i + 8] = hi[i]; }
    return f;
}

// B fragment from fp32 weights W[n][kbase + 16*hl .. +15], converted to f16.
__device__ __forceinline__ v16h load_frag_b_w(const float* p) {
    v16h f;
#pragma unroll
    for (int i = 0; i < 16; ++i) f[i] = (_Float16)p[i];
    return f;
}

// reductions across a 16-lane half (one C-tile row lives in one half)
__device__ __forceinline__ float hmax16(float v) {
#pragma unroll
    for (int m = 1; m < 16; m <<= 1) v = fmaxf(v, __shfl_xor(v, m, 16));
    return v;
}
__device__ __forceinline__ float hsum16(float v) {
#pragma unroll
    for (int m = 1; m < 16; m <<= 1) v += __shfl_xor(v, m, 16);
    return v;
}

// ---------------------------------------------------------------------------
// Kernel 1: qkv = x^T @ W^T + b ; l2-normalize q,k per (head,token); q *= SCALE
// grid = 256 (one window), block = 384 (12 waves).
// LDS: xbuf[256][392] f16 (x transposed to token-major).
// Wave w owns output n-tiles [6w, 6w+6): waves 0-3 -> q, 4-7 -> k, 8-11 -> v;
// tile pairs are head-aligned so the 32-wide L2 norm stays inside the wave.
// ---------------------------------------------------------------------------
#define XLD 392   // 384 + 8 pad (stride = 196 dwords -> bank-spread)

__global__ void __launch_bounds__(384)
k_qkv(const float* __restrict__ x, const float* __restrict__ qkv_w,
      const float* __restrict__ qkv_b,
      _Float16* __restrict__ qws, _Float16* __restrict__ kws,
      _Float16* __restrict__ vws) {
    extern __shared__ _Float16 xbuf[];   // [NTOK][XLD]
    const int b = blockIdx.x;
    const float* xb = x + (size_t)b * DIM * NTOK;   // layout (c, n)

    // transpose-fill: two c's per thread -> one b32 LDS store
    for (int i = threadIdx.x; i < (DIM / 2) * NTOK; i += 384) {
        int n = i & 255;
        int c = (i >> 8) * 2;
        v2h pr;
        pr[0] = (_Float16)xb[(size_t)c * NTOK + n];
        pr[1] = (_Float16)xb[(size_t)(c + 1) * NTOK + n];
        *(v2h*)(xbuf + n * XLD + c) = pr;
    }
    __syncthreads();

    const int lane = threadIdx.x & 31;
    const int wv   = threadIdx.x >> 5;
    const int hl   = lane >> 4;
    const int ln   = lane & 15;

    for (int p = 0; p < 3; ++p) {
        const int t0 = wv * 6 + 2 * p;       // even n-tile -> head aligned
        const int o0 = t0 * 16 + ln;         // output column (low tile)
        const int o1 = o0 + 16;

        v16h W0[12], W1[12];
#pragma unroll
        for (int kk = 0; kk < 12; ++kk) {
            W0[kk] = load_frag_b_w(qkv_w + (size_t)o0 * DIM + kk * 32 + 16 * hl);
            W1[kk] = load_frag_b_w(qkv_w + (size_t)o1 * DIM + kk * 32 + 16 * hl);
        }
        const float b0 = qkv_b[o0];
        const float b1 = qkv_b[o1];

        const int sec  = t0 / 24;                 // 0=q 1=k 2=v
        const int head = (t0 % 24) >> 1;
        _Float16* dst = (sec == 0 ? qws : (sec == 1 ? kws : vws)) +
                        ((size_t)(b * NHEAD + head)) * NTOK * HDIM;

        for (int mt = 0; mt < 16; ++mt) {
            v8f acc0 = vzero8();
            v8f acc1 = vzero8();
            const _Float16* arow = xbuf + (mt * 16 + ln) * XLD;
            v16h A = load_frag_a(arow, hl);            // kk = 0
#pragma unroll
            for (int kk = 0; kk < 12; ++kk) {
                v16h An = A;                           // pipeline next A-frag
                if (kk < 11) An = load_frag_a(arow + (kk + 1) * 32, hl);
                acc0 = wmma_f16(A, W0[kk], acc0);
                acc1 = wmma_f16(A, W1[kk], acc1);
                A = An;
            }
#pragma unroll
            for (int r = 0; r < 8; ++r) { acc0[r] += b0; acc1[r] += b1; }

            if (sec < 2) {                        // l2norm over d=32 (q and k)
#pragma unroll
                for (int r = 0; r < 8; ++r) {
                    float ss = hsum16(acc0[r] * acc0[r] + acc1[r] * acc1[r]);
                    float sc = 1.0f / fmaxf(sqrtf(ss), 1e-12f);
                    if (sec == 0) sc *= SCALE_F;  // fold attention scale into q
                    acc0[r] *= sc; acc1[r] *= sc;
                }
            }
#pragma unroll
            for (int r = 0; r < 8; ++r) {
                int tok = mt * 16 + r + 8 * hl;
                dst[tok * HDIM + ln]      = (_Float16)acc0[r];
                dst[tok * HDIM + 16 + ln] = (_Float16)acc1[r];
            }
        }
    }
}

// ---------------------------------------------------------------------------
// Kernel 2: attention per (window, head).
// grid = 3072, block = 512 (16 waves; wave w owns M-tile w => 16 score rows).
// S = q@k^T : one WMMA per 16x16 tile (K = HDIM = 32). Softmax in registers
// with 16-lane shuffle reductions.
// P@V uses a K-permuted token axis perm(tok) = (tok&15)*16 + (tok>>4) applied
// identically to P's columns and V's rows (legal: K is only summed over).
// This makes each lane's 16 softmax values contiguous -> 2x ds_store_b128.
// ---------------------------------------------------------------------------
#define PLD 264   // 256 + 8 pad halves (stride 132 dwords -> conflict-free b128)

__global__ void __launch_bounds__(512)
k_attn(const _Float16* __restrict__ qws, const _Float16* __restrict__ kws,
       const _Float16* __restrict__ vws, _Float16* __restrict__ ows) {
    extern __shared__ char smem[];
    _Float16* vT = (_Float16*)smem;                       // [HDIM][PLD] (perm'd K)
    _Float16* pb = (_Float16*)(smem + HDIM * PLD * 2);    // 16 x [16][PLD]

    const int bh = blockIdx.x;
    const int b  = bh / NHEAD;
    const int h  = bh % NHEAD;
    const _Float16* qp = qws + (size_t)bh * NTOK * HDIM;
    const _Float16* kp = kws + (size_t)bh * NTOK * HDIM;
    const _Float16* vp = vws + (size_t)bh * NTOK * HDIM;

    // V transposed + K-permuted into LDS: vT[d][perm(tok)], perm = m*16 + j
    // (tok = j*16 + m).  Pair j, j+1 -> adjacent columns -> one b32 store.
    for (int i = threadIdx.x; i < (NTOK / 2) * HDIM; i += 512) {
        int d  = i & 31;
        int t2 = i >> 5;             // 0..127
        int m  = t2 & 15;
        int j  = (t2 >> 4) * 2;      // even j
        v2h pr;
        pr[0] = vp[(j * 16 + m) * HDIM + d];
        pr[1] = vp[((j + 1) * 16 + m) * HDIM + d];
        *(v2h*)(vT + d * PLD + m * 16 + j) = pr;
    }
    __syncthreads();

    const int lane = threadIdx.x & 31;
    const int wv   = threadIdx.x >> 5;
    const int hl   = lane >> 4;
    const int ln   = lane & 15;
    const int mt   = wv;

    // ---- S = (q*scale) @ k^T ------------------------------------------------
    const v16h qa = load_frag_a(qp + (size_t)(mt * 16 + ln) * HDIM, hl);
    v8f S[16];
    {
        v16h kb = load_frag_b(kp + (size_t)ln * HDIM + 16 * hl);   // j = 0
#pragma unroll
        for (int j = 0; j < 16; ++j) {
            v16h kbn = kb;                                          // pipeline
            if (j < 15)
                kbn = load_frag_b(kp + (size_t)((j + 1) * 16 + ln) * HDIM + 16 * hl);
            S[j] = wmma_f16(qa, kb, vzero8());
            kb = kbn;
        }
    }

    // ---- softmax + clip -> P (f16, per-wave LDS, K-permuted columns) -------
    _Float16* pw = pb + (size_t)wv * 16 * PLD;
#pragma unroll
    for (int r = 0; r < 8; ++r) {
        float mx = S[0][r];
#pragma unroll
        for (int j = 1; j < 16; ++j) mx = fmaxf(mx, S[j][r]);
        mx = hmax16(mx);
        float sum = 0.0f;
#pragma unroll
        for (int j = 0; j < 16; ++j) {
            float e = __expf(S[j][r] - mx);
            S[j][r] = e;
            sum += e;
        }
        sum = hsum16(sum);
        const float inv = 1.0f / sum;
        const int row = r + 8 * hl;
        // this lane's columns are toks {j*16+ln} -> perm = ln*16 + j : contiguous
        v8h lo, hi;
#pragma unroll
        for (int j = 0; j < 8; ++j) {
            lo[j] = (_Float16)fminf(fmaxf(S[j][r]     * inv, 1e-6f), 1.0f);
            hi[j] = (_Float16)fminf(fmaxf(S[j + 8][r] * inv, 1e-6f), 1.0f);
        }
        *(v8h*)(pw + row * PLD + ln * 16)     = lo;
        *(v8h*)(pw + row * PLD + ln * 16 + 8) = hi;
    }
    __syncthreads();

    // ---- O = P @ V  (K axis permuted consistently in P and vT) -------------
    const _Float16* arow = pw + ln * PLD;     // A row m = ln
    v8f O0 = vzero8(), O1 = vzero8();
    {
        v16h pa = load_frag_a(arow, hl);
        v16h v0 = load_frag_b(vT + ln * PLD        + 16 * hl);
        v16h v1 = load_frag_b(vT + (16 + ln) * PLD + 16 * hl);
#pragma unroll
        for (int kk = 0; kk < 8; ++kk) {
            v16h pan = pa, v0n = v0, v1n = v1;
            if (kk < 7) {                                         // pipeline
                pan = load_frag_a(arow + (kk + 1) * 32, hl);
                v0n = load_frag_b(vT + ln * PLD        + (kk + 1) * 32 + 16 * hl);
                v1n = load_frag_b(vT + (16 + ln) * PLD + (kk + 1) * 32 + 16 * hl);
            }
            O0 = wmma_f16(pa, v0, O0);
            O1 = wmma_f16(pa, v1, O1);
            pa = pan; v0 = v0n; v1 = v1n;
        }
    }

    // attn-out ws layout: [b][token][c = h*32 + d]  (token-major, c contiguous)
    _Float16* op = ows + (size_t)b * NTOK * DIM + h * HDIM;
#pragma unroll
    for (int r = 0; r < 8; ++r) {
        int tok = mt * 16 + r + 8 * hl;
        op[(size_t)tok * DIM + ln]      = (_Float16)O0[r];
        op[(size_t)tok * DIM + 16 + ln] = (_Float16)O1[r];
    }
}

// ---------------------------------------------------------------------------
// Kernel 3: out = attn_out @ proj_w^T + proj_b, stored transposed (B,C,H,W).
// grid = 256, block = 384 (12 waves, 2 output n-tiles each).
// ---------------------------------------------------------------------------
__global__ void __launch_bounds__(384)
k_proj(const _Float16* __restrict__ ows, const float* __restrict__ proj_w,
       const float* __restrict__ proj_b, float* __restrict__ out) {
    const int b = blockIdx.x;
    const _Float16* ap = ows + (size_t)b * NTOK * DIM;
    const int lane = threadIdx.x & 31;
    const int wv   = threadIdx.x >> 5;
    const int hl   = lane >> 4;
    const int ln   = lane & 15;

    for (int t = 0; t < 2; ++t) {
        const int nt = wv * 2 + t;
        const int o  = nt * 16 + ln;

        v16h W[12];
#pragma unroll
        for (int kk = 0; kk < 12; ++kk)
            W[kk] = load_frag_b_w(proj_w + (size_t)o * DIM + kk * 32 + 16 * hl);
        const float bias = proj_b[o];
        float* outp = out + (size_t)b * DIM * NTOK + (size_t)o * NTOK;

        for (int mt = 0; mt < 16; ++mt) {
            v8f acc = vzero8();
            const _Float16* arow = ap + (size_t)(mt * 16 + ln) * DIM;
            v16h A = load_frag_a(arow, hl);
#pragma unroll
            for (int kk = 0; kk < 12; ++kk) {
                v16h An = A;                          // pipeline next A-frag
                if (kk < 11) An = load_frag_a(arow + (kk + 1) * 32, hl);
                acc = wmma_f16(A, W[kk], acc);
                A = An;
            }
            const int tok0 = mt * 16 + 8 * hl;
#pragma unroll
            for (int r = 0; r < 8; ++r) outp[tok0 + r] = acc[r] + bias;
        }
    }
}

// ---------------------------------------------------------------------------
extern "C" void kernel_launch(void* const* d_in, const int* in_sizes, int n_in,
                              void* d_out, int out_size, void* d_ws, size_t ws_size,
                              hipStream_t stream) {
    (void)in_sizes; (void)n_in; (void)out_size; (void)ws_size;
    const float* x      = (const float*)d_in[0];
    const float* qkv_w  = (const float*)d_in[1];
    const float* qkv_b  = (const float*)d_in[2];
    const float* proj_w = (const float*)d_in[3];
    const float* proj_b = (const float*)d_in[4];
    float* out = (float*)d_out;

    // ws: q | k | v | attn_out, each B*H*N*D halves (48 MiB) -> 192 MiB total
    const size_t per = (size_t)B_WIN * NHEAD * NTOK * HDIM;
    _Float16* qws = (_Float16*)d_ws;
    _Float16* kws = qws + per;
    _Float16* vws = kws + per;
    _Float16* ows = vws + per;

    const size_t lds1 = (size_t)NTOK * XLD * sizeof(_Float16);              // 200704 B
    const size_t lds2 = (size_t)(HDIM + 16 * 16) * PLD * sizeof(_Float16);  // 152064 B

    k_qkv <<<B_WIN,        384, lds1, stream>>>(x, qkv_w, qkv_b, qws, kws, vws);
    k_attn<<<B_WIN * NHEAD, 512, lds2, stream>>>(qws, kws, vws, ows);
    k_proj<<<B_WIN,        384, 0,    stream>>>(ows, proj_w, proj_b, out);
}
